// ReferenceUpdater_163208757444
// MI455X (gfx1250) — matrix operand order repaired and checked by reading, not compiled, not run on hardware
//
#include <hip/hip_runtime.h>

typedef __attribute__((ext_vector_type(16))) __bf16 v16bf;
typedef __attribute__((ext_vector_type(8)))  float  v8f;

#define Dm 256
#define Hm 1024
#define TM 64          // rows per workgroup
#define ROW1 264       // ctx/gate LDS row stride (bf16), 528B = 33*16B (odd 16B units)
#define ROWH 1032      // h LDS row stride (bf16), 2064B = 129*16B
#define ROWY 260       // y LDS row stride (f32)

// ---------------- weight pack: fp32 -> bf16 ----------------
__global__ void pack_weights(const float* __restrict__ w_in,
                             const float* __restrict__ w_gate,
                             const float* __restrict__ w_out,
                             __bf16* __restrict__ dst) {
    int i = blockIdx.x * blockDim.x + threadIdx.x;   // 0 .. 786431
    float v;
    if (i < 262144)       v = w_in[i];
    else if (i < 524288)  v = w_gate[i - 262144];
    else                  v = w_out[i - 524288];
    dst[i] = (__bf16)v;
}

// ---------------- pass-through copy of unmasked rows ----------------
__global__ void copy_pass(const float4* __restrict__ in, float4* __restrict__ out) {
    long i = (long)blockIdx.x * blockDim.x + threadIdx.x;  // float4 index
    int row = (int)(i >> 6);                               // 64 float4 per 256-elem row
    if ((row & 3) != 0) out[i] = in[i];                    // masked rows written by ffn kernel
}

// ---------------- fused gather + gated FFN + layernorm + scatter ----------------
__global__ __launch_bounds__(256, 1)
void ffn_kernel(const float* __restrict__ tok,
                const float* __restrict__ ref,
                const __bf16* __restrict__ w_in_bf,
                const __bf16* __restrict__ w_gate_bf,
                const __bf16* __restrict__ w_out_bf,
                const float* __restrict__ bias_in,
                const float* __restrict__ bias_gate,
                const float* __restrict__ bias_out,
                const float* __restrict__ ln_g,
                const float* __restrict__ ln_b,
                const int*   __restrict__ ids,
                float* __restrict__ out) {
    __shared__ __align__(16) __bf16 s_ctx[TM * ROW1];
    __shared__ __align__(16) __bf16 s_gate[TM * ROW1];
    __shared__ __align__(16) __bf16 s_h[TM * ROWH];
    __shared__ __align__(16) float  s_y[TM * ROWY];
    __shared__ float s_red[TM * 4 * 2];
    __shared__ float s_stats[TM * 2];

    const int tid  = threadIdx.x;
    const int lane = tid & 31;
    const int wave = tid >> 5;
    const int l16  = lane & 15;
    const int half = lane >> 4;
    const int g0   = blockIdx.x * TM;

    // ---- stage ctx (gather) and gate rows into LDS as bf16 ----
    {
        const int r   = tid >> 2;      // 0..63
        const int sub = tid & 3;       // 64-elem chunk
        const int g   = g0 + r;
        const long id = (long)ids[g];
        const float* cp = ref + id * (long)Dm + sub * 64;
        const float* gp = tok + (long)(4 * g) * Dm + sub * 64;
        __bf16* cd = s_ctx  + r * ROW1 + sub * 64;
        __bf16* gd = s_gate + r * ROW1 + sub * 64;
#pragma unroll
        for (int c = 0; c < 64; ++c) {
            cd[c] = (__bf16)cp[c];
            gd[c] = (__bf16)gp[c];
        }
    }
    __syncthreads();

    // ---- GEMM1: h = (ctx@w_in + b_in) * silu(gate@w_gate + b_gate) ----
    for (int t = 0; t < 8; ++t) {
        const int n0 = (wave * 8 + t) * 16;
        v8f ai[4], ag[4];
#pragma unroll
        for (int m = 0; m < 4; ++m) {
#pragma unroll
            for (int v = 0; v < 8; ++v) { ai[m][v] = 0.0f; ag[m][v] = 0.0f; }
        }
#pragma unroll
        for (int ks = 0; ks < 8; ++ks) {
            const int krow = ks * 32 + half * 8;   // B: lane <-> N column, K stripes with half-lane split
            const __bf16* bpi = w_in_bf   + (size_t)krow * Hm + n0 + l16;
            const __bf16* bpg = w_gate_bf + (size_t)krow * Hm + n0 + l16;
            v16bf bi, bg;
#pragma unroll
            for (int j = 0; j < 8; ++j) { bi[j] = bpi[j * Hm]; bi[8 + j] = bpi[(16 + j) * Hm]; bg[j] = bpg[j * Hm]; bg[8 + j] = bpg[(16 + j) * Hm]; }
            const int kA = ks * 32 + half * 8;            // A: K stripes with half-lane split
#pragma unroll
            for (int m = 0; m < 4; ++m) {
                const __bf16* ap = s_ctx + (m * 16 + l16) * ROW1 + kA;
                v16bf a;
#pragma unroll
                for (int j = 0; j < 8; ++j) { a[j] = ap[j]; a[8 + j] = ap[16 + j]; }
                ai[m] = __builtin_amdgcn_wmma_f32_16x16x32_bf16(
                            false, a, false, bi, (short)0, ai[m], false, false);
                const __bf16* ap2 = s_gate + (m * 16 + l16) * ROW1 + kA;
                v16bf a2;
#pragma unroll
                for (int j = 0; j < 8; ++j) { a2[j] = ap2[j]; a2[8 + j] = ap2[16 + j]; }
                ag[m] = __builtin_amdgcn_wmma_f32_16x16x32_bf16(
                            false, a2, false, bg, (short)0, ag[m], false, false);
            }
        }
        const float bin_v = bias_in[n0 + l16];
        const float bgt_v = bias_gate[n0 + l16];
#pragma unroll
        for (int m = 0; m < 4; ++m) {
#pragma unroll
            for (int v = 0; v < 8; ++v) {
                const float gg = ag[m][v] + bgt_v;
                const float s  = gg / (1.0f + __expf(-gg));       // silu
                const float hv = (ai[m][v] + bin_v) * s;
                const int row  = m * 16 + v + 8 * half;           // C layout: v+8*half
                s_h[row * ROWH + n0 + l16] = (__bf16)hv;
            }
        }
    }
    __syncthreads();

    // ---- GEMM2: y = h @ w_out (+ b_out) ----
    for (int t = 0; t < 2; ++t) {
        const int n0 = (wave * 2 + t) * 16;
        v8f acc[4];
#pragma unroll
        for (int m = 0; m < 4; ++m)
#pragma unroll
            for (int v = 0; v < 8; ++v) acc[m][v] = 0.0f;

        for (int ks = 0; ks < 32; ++ks) {
            const int krow = ks * 32 + half * 8;
            const __bf16* bp = w_out_bf + (size_t)krow * Dm + n0 + l16;
            v16bf b;
#pragma unroll
            for (int j = 0; j < 8; ++j) { b[j] = bp[j * Dm]; b[8 + j] = bp[(16 + j) * Dm]; }
            const int kA = ks * 32 + half * 8;
#pragma unroll
            for (int m = 0; m < 4; ++m) {
                const __bf16* ap = s_h + (m * 16 + l16) * ROWH + kA;
                v16bf a;
#pragma unroll
                for (int j = 0; j < 8; ++j) { a[j] = ap[j]; a[8 + j] = ap[16 + j]; }
                acc[m] = __builtin_amdgcn_wmma_f32_16x16x32_bf16(
                             false, a, false, b, (short)0, acc[m], false, false);
            }
        }
        const float bo = bias_out[n0 + l16];
#pragma unroll
        for (int m = 0; m < 4; ++m)
#pragma unroll
            for (int v = 0; v < 8; ++v)
                s_y[(m * 16 + v + 8 * half) * ROWY + n0 + l16] = acc[m][v] + bo;
    }
    __syncthreads();

    // ---- epilogue: residual (re-gather ctx fp32) + layernorm + scatter ----
    {
        const int r   = tid >> 2;
        const int sub = tid & 3;
        const int g   = g0 + r;
        const long id = (long)ids[g];
        const float* cp = ref + id * (long)Dm + sub * 64;
        float* yr = s_y + r * ROWY + sub * 64;
        float s = 0.0f, ss = 0.0f;
#pragma unroll
        for (int c = 0; c < 64; ++c) {
            const float v = yr[c] + cp[c];
            yr[c] = v;
            s += v; ss += v * v;
        }
        s_red[(r * 4 + sub) * 2 + 0] = s;
        s_red[(r * 4 + sub) * 2 + 1] = ss;
    }
    __syncthreads();
    {
        const int r   = tid >> 2;
        const int sub = tid & 3;
        if (sub == 0) {
            float S = 0.0f, SS = 0.0f;
#pragma unroll
            for (int k = 0; k < 4; ++k) {
                S  += s_red[(r * 4 + k) * 2 + 0];
                SS += s_red[(r * 4 + k) * 2 + 1];
            }
            const float mu  = S * (1.0f / 256.0f);
            const float var = SS * (1.0f / 256.0f) - mu * mu;
            s_stats[r * 2 + 0] = mu;
            s_stats[r * 2 + 1] = rsqrtf(var + 1e-5f);
        }
    }
    __syncthreads();
    {
        const int r   = tid >> 2;
        const int sub = tid & 3;
        const int g   = g0 + r;
        const float mu = s_stats[r * 2 + 0];
        const float rs = s_stats[r * 2 + 1];
        const float* yr = s_y + r * ROWY + sub * 64;
        const float* gv = ln_g + sub * 64;
        const float* bv = ln_b + sub * 64;
        float* op = out + (size_t)(4 * g) * Dm + sub * 64;
#pragma unroll
        for (int c = 0; c < 64; ++c)
            op[c] = (yr[c] - mu) * rs * gv[c] + bv[c];
    }
}

extern "C" void kernel_launch(void* const* d_in, const int* in_sizes, int n_in,
                              void* d_out, int out_size, void* d_ws, size_t ws_size,
                              hipStream_t stream) {
    const float* tok    = (const float*)d_in[0];
    const float* ref    = (const float*)d_in[1];
    const float* w_in   = (const float*)d_in[2];
    const float* b_in   = (const float*)d_in[3];
    const float* w_gate = (const float*)d_in[4];
    const float* b_gate = (const float*)d_in[5];
    const float* w_out  = (const float*)d_in[6];
    const float* b_out  = (const float*)d_in[7];
    const float* ln_g   = (const float*)d_in[8];
    const float* ln_b   = (const float*)d_in[9];
    const int*   ids    = (const int*)d_in[11];
    float* out = (float*)d_out;
    __bf16* wsb = (__bf16*)d_ws;   // [w_in_bf | w_gate_bf | w_out_bf], 262144 elems each

    // 1) pack weights to bf16 in workspace
    pack_weights<<<3072, 256, 0, stream>>>(w_in, w_gate, w_out, wsb);
    // 2) pass-through copy for unmasked rows (3/4 of tensor)
    copy_pass<<<65536, 256, 0, stream>>>((const float4*)tok, (float4*)out);
    // 3) fused gather + FFN + layernorm + scatter for the 65536 masked rows
    ffn_kernel<<<1024, 256, 0, stream>>>(tok, ref,
                                         wsb, wsb + 262144, wsb + 524288,
                                         b_in, b_gate, b_out, ln_g, ln_b,
                                         ids, out);
}
